// GraphAttentionLayer_11046655886013
// MI455X (gfx1250) — matrix operand order repaired and checked by reading, not compiled, not run on hardware
//
#include <hip/hip_runtime.h>
#include <hip/hip_bf16.h>
#include <stdint.h>

// ---------------------------------------------------------------------------
// GAT layer, MI455X (gfx1250, wave32, WMMA + TDM).
// Sizes fixed by the reference: M=32, N=1024, Fin=Fout=128.
//
// h'[m,i,:] = ELU( sum_j E[m,i,j] * Whn[m,j,:] )
//   E[i,j]  = adj>0 ? exp(lrelu(e_src[i]+e_dst[j]) - colmax[j]) : 0
//   WhnT    = bf16( Wh / colsum[j] ) stored o-major for TDM tile loads
// ---------------------------------------------------------------------------

#define Mb   32
#define Nn   1024
#define Ff   128
#define ALPHA 0.2f
#define NEGINF (-9.0e15f)
#define LDST 34   // padded LDS row stride (elements) -> conflict-free frag reads

typedef __attribute__((ext_vector_type(16))) __bf16 v16bf;
typedef __attribute__((ext_vector_type(8)))  float  v8f;
typedef __attribute__((ext_vector_type(4)))  unsigned int v4u;
typedef __attribute__((ext_vector_type(8)))  int  v8i;
typedef __attribute__((ext_vector_type(4)))  int  v4i;

union BFrag { v16bf v; unsigned int u[8]; };

__device__ __forceinline__ unsigned short f2bf(float x) {
    unsigned int u = __float_as_uint(x);
    unsigned int r = u + 0x7FFFu + ((u >> 16) & 1u);   // round-to-nearest-even
    return (unsigned short)(r >> 16);
}
__device__ __forceinline__ unsigned int pack2bf(float a, float b) {
    return (unsigned int)f2bf(a) | ((unsigned int)f2bf(b) << 16);
}

// A matrix 16x32 bf16 fragment (ISA 7.12.2): lane M=lane&15;
// v0..3: K=2v, v4..7: K=2v+8, +8 for lanes 16..31. Pairs packed per VGPR.
__device__ __forceinline__ v16bf make_afrag(const unsigned short* tile, int lane) {
    int mrow = lane & 15, hi = lane >> 4;
    BFrag f;
#pragma unroll
    for (int v = 0; v < 8; ++v) {
        int k = (v < 4 ? 2 * v : 2 * v + 8) + (hi ? 8 : 0);
        f.u[v] = *(const unsigned int*)(tile + mrow * LDST + k);
    }
    return f.v;
}

// B matrix 32x16 bf16 fragment: N=lane&15; v0..3: K=2v, v4..7: K=2v+8, +8 for lanes 16..31.
// bt is a [128][LDST] (o-major, k-minor) transposed tile.
__device__ __forceinline__ v16bf make_bfrag(const unsigned short* bt, int otile, int lane) {
    int n = lane & 15, hi = lane >> 4;
    int o = otile * 16 + n;
    BFrag f;
#pragma unroll
    for (int v = 0; v < 8; ++v) {
        int k = (v < 4 ? 2 * v : 2 * v + 8) + (hi ? 8 : 0);
        f.u[v] = *(const unsigned int*)(bt + o * LDST + k);
    }
    return f.v;
}

__device__ __forceinline__ v8f wmma_bf16(v16bf a, v16bf b, v8f c) {
    return __builtin_amdgcn_wmma_f32_16x16x32_bf16(false, a, false, b, (short)0, c,
                                                   false, false);
}

#if __has_builtin(__builtin_amdgcn_tensor_load_to_lds) && \
    __has_builtin(__builtin_amdgcn_s_wait_tensorcnt)
#define USE_TDM 1
#else
#define USE_TDM 0
#endif

#if USE_TDM
// TDM: load a [128 rows x 64B] 2D tile (bf16, row stride 2KB in memory) into
// LDS with 4B padding after every 64B row  ->  LDS row stride 68B == LDST(34).
__device__ __forceinline__ void tdm_load_bt(unsigned int lds_off, const void* gptr) {
    unsigned long long ga = (unsigned long long)gptr;
    v4u g0 = { 1u,                                     // count=1 (valid, user)
               lds_off,                                // lds_addr
               (unsigned int)ga,                       // global_addr[31:0]
               ((unsigned int)(ga >> 32) & 0x01FFFFFFu) | (2u << 30) }; // type=2
    v8i g1 = { (int)((1u << 16) | (1u << 20) | (3u << 22)), // 2B elems, pad 1DW/16DW
               (int)(1024u << 16),                     // tensor_dim0 = 1024
               (int)(128u << 16),                      // tensor_dim1 = 128
               (int)(32u << 16),                       // tile_dim0 = 32
               128,                                    // tile_dim1 = 128
               1024,                                   // tensor_dim0_stride = 1024
               0, 0 };
    v4i z4 = { 0, 0, 0, 0 };
    v8i z8 = { 0, 0, 0, 0, 0, 0, 0, 0 };
    // clang-23 / therock lane: 6-arg form (extra int32x8 group before cpol)
    __builtin_amdgcn_tensor_load_to_lds(g0, g1, z4, z4, z8, 0);
}
#endif

// ---------------------------------------------------------------------------
// Kernel 1: Wh = h @ W  (bf16 WMMA, fp32 accum)  + e_src/e_dst from accum regs
// grid (8 iblk, 32 m) x 256
// ---------------------------------------------------------------------------
__global__ void k_wh(const float* __restrict__ h, const float* __restrict__ W,
                     const float* __restrict__ a, float* __restrict__ Wh,
                     float* __restrict__ esrc, float* __restrict__ edst) {
    __shared__ unsigned short As[128 * LDST];
    __shared__ unsigned short Bt[128 * LDST];

    const int m = blockIdx.y, i0 = blockIdx.x * 128;
    const int t = threadIdx.x, lane = t & 31, wave = t >> 5;

    v8f acc[8];
#pragma unroll
    for (int ot = 0; ot < 8; ++ot) acc[ot] = (v8f){};

    for (int k0 = 0; k0 < Ff; k0 += 32) {
        __syncthreads();
#pragma unroll
        for (int c = 0; c < 8; ++c) {            // h tile [128][32] -> bf16 pairs
            int f = c * 256 + t, kp = f & 15, ii = f >> 4;
            float2 hx = *(const float2*)&h[(size_t)(m * Nn + i0 + ii) * Ff + k0 + 2 * kp];
            *(unsigned int*)(As + ii * LDST + 2 * kp) = pack2bf(hx.x, hx.y);
        }
#pragma unroll
        for (int c = 0; c < 16; ++c) {           // W tile [32][128] -> Bt[o][k]
            int f = c * 256 + t, o = f & 127, kk = f >> 7;
            Bt[o * LDST + kk] = f2bf(W[(size_t)(k0 + kk) * Ff + o]);
        }
        __syncthreads();
        v16bf af = make_afrag(As + wave * 16 * LDST, lane);
        v16bf bf[8];
#pragma unroll
        for (int ot = 0; ot < 8; ++ot) bf[ot] = make_bfrag(Bt, ot, lane); // preload
#pragma unroll
        for (int ot = 0; ot < 8; ++ot)           // back-to-back WMMAs
            acc[ot] = wmma_bf16(af, bf[ot], acc[ot]);
    }

    // C/D layout: lane holds o=(lane&15)+16*ot, row i = r + 8*(lane>>4).
    const int hi = lane >> 4, n = lane & 15;
#pragma unroll
    for (int r = 0; r < 8; ++r) {
        int i = i0 + wave * 16 + r + 8 * hi;
        float ssrc = 0.f, sdst = 0.f;
#pragma unroll
        for (int ot = 0; ot < 8; ++ot) {
            int o = ot * 16 + n;
            float v = acc[ot][r];
            Wh[(size_t)(m * Nn + i) * Ff + o] = v;
            ssrc += v * a[o];
            sdst += v * a[Ff + o];
        }
#pragma unroll
        for (int msk = 1; msk < 16; msk <<= 1) {    // reduce over the 16 o-lanes
            ssrc += __shfl_xor(ssrc, msk, 32);
            sdst += __shfl_xor(sdst, msk, 32);
        }
        if (n == 0) {
            esrc[m * Nn + i] = ssrc;
            edst[m * Nn + i] = sdst;
        }
    }
}

// ---------------------------------------------------------------------------
// Kernel 2a: per-column (m,j) online (max,sum) over a 128-row i chunk
// grid (4 jblk, 8 ichunk, 32 m) x 256
// ---------------------------------------------------------------------------
__global__ void k_colpart(const int* __restrict__ adj, const float* __restrict__ esrc,
                          const float* __restrict__ edst, float* __restrict__ pmax,
                          float* __restrict__ psum) {
    const int t = threadIdx.x;
    const int m = blockIdx.z, ic = blockIdx.y;
    const int j = blockIdx.x * 256 + t;
    const float ed = edst[m * Nn + j];

    float mx = NEGINF, sum = 0.f;
    const int ibeg = ic * 128;
#pragma unroll 4
    for (int i = ibeg; i < ibeg + 128; ++i) {
        int ad = adj[(size_t)(m * Nn + i) * Nn + j];       // coalesced along j
        float e = esrc[m * Nn + i] + ed;                   // scalar broadcast
        e = e > 0.f ? e : ALPHA * e;
        float v = (ad > 0) ? e : NEGINF;
        float nm = fmaxf(mx, v);
        sum = sum * __expf(mx - nm) + __expf(v - nm);
        mx = nm;
    }
    pmax[(size_t)(ic * Mb + m) * Nn + j] = mx;
    psum[(size_t)(ic * Mb + m) * Nn + j] = sum;
}

// Kernel 2b: combine the 8 i-chunks -> colmax/colsum.  grid 128 x 256
__global__ void k_colcombine(const float* __restrict__ pmax, const float* __restrict__ psum,
                             float* __restrict__ colmax, float* __restrict__ colsum) {
    const int col = blockIdx.x * 256 + threadIdx.x;        // m*1024 + j
    float gm = NEGINF;
#pragma unroll
    for (int ic = 0; ic < 8; ++ic) gm = fmaxf(gm, pmax[(size_t)ic * Mb * Nn + col]);
    float s = 0.f;
#pragma unroll
    for (int ic = 0; ic < 8; ++ic)
        s += psum[(size_t)ic * Mb * Nn + col] * __expf(pmax[(size_t)ic * Mb * Nn + col] - gm);
    colmax[col] = gm;
    colsum[col] = s;
}

// ---------------------------------------------------------------------------
// Kernel 2c: WhnT[m][o][j] = bf16(Wh[m][j][o] / colsum[m][j])  (transposed for
// TDM tile loads).  LDS transpose; coalesced reads.  grid (32 jblk, 32 m) x 256
// ---------------------------------------------------------------------------
__global__ void k_whnT(const float* __restrict__ Wh, const float* __restrict__ colsum,
                       unsigned short* __restrict__ WhnT) {
    __shared__ unsigned short T[128 * LDST];               // [o][j] tile
    const int m = blockIdx.y, j0 = blockIdx.x * 32;
    const int t = threadIdx.x;
#pragma unroll
    for (int c = 0; c < 16; ++c) {
        int f = c * 256 + t, o = f & 127, jj = f >> 7;
        float inv = 1.0f / colsum[m * Nn + j0 + jj];
        T[o * LDST + jj] = f2bf(Wh[(size_t)(m * Nn + j0 + jj) * Ff + o] * inv);
    }
    __syncthreads();
    unsigned int* dst = (unsigned int*)WhnT;
#pragma unroll
    for (int c = 0; c < 8; ++c) {
        int f = c * 256 + t, op = f & 15, o = f >> 4;
        unsigned int v = *(const unsigned int*)(T + o * LDST + 2 * op);
        dst[((size_t)(m * Ff + o) * Nn + j0) / 2 + op] = v;
    }
}

// ---------------------------------------------------------------------------
// Kernel 3: h' = ELU( E @ Whn ).  E tiles built on the fly; B tiles streamed
// into LDS by the Tensor Data Mover (overlapped with E construction).
// grid (8 iblk, 32 m) x 256 (8 waves, 16 i-rows each).
// ---------------------------------------------------------------------------
__global__ void k_main(const int* __restrict__ adj, const float* __restrict__ esrc,
                       const float* __restrict__ edst, const float* __restrict__ colmax,
                       const unsigned short* __restrict__ WhnT, float* __restrict__ out) {
    __shared__ unsigned short Es[128 * LDST];
    __shared__ unsigned short Bt[128 * LDST];

    const int m = blockIdx.y, i0 = blockIdx.x * 128;
    const int t = threadIdx.x, lane = t & 31, wave = t >> 5;

    v8f acc[8];
#pragma unroll
    for (int ot = 0; ot < 8; ++ot) acc[ot] = (v8f){};

    for (int j0 = 0; j0 < Nn; j0 += 32) {
        __syncthreads();                                   // LDS safe to overwrite
#if USE_TDM
        if (wave == 0)                                     // DMA B tile, async
            tdm_load_bt((unsigned int)(unsigned long long)(void*)Bt,
                        WhnT + (size_t)m * Ff * Nn + j0);
#endif
        if (j0 + 32 < Nn)                                  // global_prefetch_b8
            __builtin_prefetch(&adj[(size_t)(m * Nn + i0 + wave * 16) * Nn + j0 + 32 + lane], 0, 1);
#pragma unroll
        for (int c = 0; c < 8; ++c) {                      // E tile (overlaps DMA)
            int f = c * 256 + t, kp = f & 15, ii = f >> 4;
            int i = i0 + ii, j = j0 + 2 * kp;
            int2 ad = *(const int2*)&adj[(size_t)(m * Nn + i) * Nn + j];
            float es = esrc[m * Nn + i];
            float e0 = es + edst[m * Nn + j],     e1 = es + edst[m * Nn + j + 1];
            e0 = e0 > 0.f ? e0 : ALPHA * e0;      e1 = e1 > 0.f ? e1 : ALPHA * e1;
            float E0 = (ad.x > 0) ? __expf(e0 - colmax[m * Nn + j])     : 0.0f;
            float E1 = (ad.y > 0) ? __expf(e1 - colmax[m * Nn + j + 1]) : 0.0f;
            *(unsigned int*)(Es + ii * LDST + 2 * kp) = pack2bf(E0, E1);
        }
#if USE_TDM
        if (wave == 0) __builtin_amdgcn_s_wait_tensorcnt(0);
#else
        {   // manual B tile: WhnT rows -> Bt[o][k] (u32 stores, conflict-free)
            const unsigned int* src = (const unsigned int*)WhnT;
#pragma unroll
            for (int c = 0; c < 8; ++c) {
                int f = c * 256 + t, op = f & 15, o = f >> 4;
                unsigned int v = src[((size_t)(m * Ff + o) * Nn + j0) / 2 + op];
                *(unsigned int*)(Bt + o * LDST + 2 * op) = v;
            }
        }
#endif
        __syncthreads();
        v16bf af = make_afrag(Es + wave * 16 * LDST, lane);
        v16bf bf[8];
#pragma unroll
        for (int ot = 0; ot < 8; ++ot) bf[ot] = make_bfrag(Bt, ot, lane); // preload
#pragma unroll
        for (int ot = 0; ot < 8; ++ot)                     // back-to-back WMMAs
            acc[ot] = wmma_bf16(af, bf[ot], acc[ot]);
    }

    const int hi = lane >> 4, n = lane & 15;
#pragma unroll
    for (int r = 0; r < 8; ++r) {
        int i = i0 + wave * 16 + r + 8 * hi;
#pragma unroll
        for (int ot = 0; ot < 8; ++ot) {
            int o = ot * 16 + n;
            float x = acc[ot][r];
            out[(size_t)(m * Nn + i) * Ff + o] = x > 0.f ? x : (__expf(x) - 1.0f); // ELU
        }
    }
}

// ---------------------------------------------------------------------------
extern "C" void kernel_launch(void* const* d_in, const int* in_sizes, int n_in,
                              void* d_out, int out_size, void* d_ws, size_t ws_size,
                              hipStream_t stream) {
    (void)in_sizes; (void)n_in; (void)out_size; (void)ws_size;
    const float* h   = (const float*)d_in[0];
    const int*   adj = (const int*)  d_in[1];
    const float* W   = (const float*)d_in[2];
    const float* a   = (const float*)d_in[3];
    float* out = (float*)d_out;

    // workspace carve-up (~26.5 MB)
    float* Wh     = (float*)d_ws;                 // 32*1024*128 f  (16 MB)
    float* esrc   = Wh + (size_t)Mb * Nn * Ff;    // 32768 f
    float* edst   = esrc + Mb * Nn;
    float* colmax = edst + Mb * Nn;
    float* colsum = colmax + Mb * Nn;
    float* pmax   = colsum + Mb * Nn;             // 8*32768 f (1 MB)
    float* psum   = pmax + 8 * Mb * Nn;
    unsigned short* WhnT = (unsigned short*)(psum + 8 * Mb * Nn); // bf16, o-major (8 MB)

    k_wh        <<<dim3(8, Mb),    256, 0, stream>>>(h, W, a, Wh, esrc, edst);
    k_colpart   <<<dim3(4, 8, Mb), 256, 0, stream>>>(adj, esrc, edst, pmax, psum);
    k_colcombine<<<128,            256, 0, stream>>>(pmax, psum, colmax, colsum);
    k_whnT      <<<dim3(32, Mb),   256, 0, stream>>>(Wh, colsum, WhnT);
    k_main      <<<dim3(8, Mb),    256, 0, stream>>>(adj, esrc, edst, colmax, WhnT, out);
}
